// Encoder_90589450207915
// MI455X (gfx1250) — matrix operand order repaired and checked by reading, not compiled, not run on hardware
//
#include <hip/hip_runtime.h>
#include <hip/hip_bf16.h>
#include <stdint.h>

// ---------------- WMMA types (CDNA5 / gfx1250, wave32) ----------------
typedef __attribute__((ext_vector_type(16))) _Float16 v16h;
typedef __attribute__((ext_vector_type(8)))  float    v8f;

union FragH {
    v16h v;
    uint4 q[2];
    _Float16 e[16];
};

// Issue one per-lane async global->LDS 16-byte copy (tracked by ASYNCcnt).
__device__ __forceinline__ void async_copy_b128(unsigned lds_off, const void* gptr) {
    asm volatile("global_load_async_to_lds_b128 %0, %1, off"
                 :: "v"(lds_off), "v"((unsigned long long)(uintptr_t)gptr)
                 : "memory");
}
__device__ __forceinline__ void wait_async0() {
    asm volatile("s_wait_asynccnt 0x0" ::: "memory");
}

// ---------------- graph-norm kernels ----------------
__global__ void k_deg_init(float* deg, int N) {
    int i = blockIdx.x * blockDim.x + threadIdx.x;
    if (i < N) deg[i] = 1.0f;            // self-loop weight 1
}

__global__ void k_deg_acc(const int* __restrict__ dst, const float* __restrict__ w,
                          float* deg, int E) {
    int e = blockIdx.x * blockDim.x + threadIdx.x;
    if (e < E) atomicAdd(&deg[dst[e]], w[e]);
}

__global__ void k_dinv(float* deg, int N) {
    int i = blockIdx.x * blockDim.x + threadIdx.x;
    if (i < N) {
        float d = deg[i];
        deg[i] = (d > 0.0f) ? rsqrtf(d) : 0.0f;
    }
}

__global__ void k_norm(const int* __restrict__ src, const int* __restrict__ dst,
                       const float* __restrict__ w, const float* __restrict__ dinv,
                       float* __restrict__ norm, int E) {
    int e = blockIdx.x * blockDim.x + threadIdx.x;
    if (e < E) norm[e] = dinv[src[e]] * w[e] * dinv[dst[e]];
}

// ---------------- conversion kernels ----------------
__global__ void k_f32_to_f16_v4(const float* __restrict__ in, _Float16* __restrict__ out,
                                size_t n4) {
    size_t i = (size_t)blockIdx.x * blockDim.x + threadIdx.x;
    if (i < n4) {
        float4 v = reinterpret_cast<const float4*>(in)[i];
        _Float16* o = out + i * 4;
        o[0] = (_Float16)v.x; o[1] = (_Float16)v.y;
        o[2] = (_Float16)v.z; o[3] = (_Float16)v.w;
    }
}

// Wt[n*K + k] = (f16) W[k*128 + n]   (transpose so B-fragment loads are contiguous in k)
__global__ void k_w_prep(const float* __restrict__ W, _Float16* __restrict__ Wt, int K) {
    int idx = blockIdx.x * blockDim.x + threadIdx.x;   // over K*128
    if (idx < K * 128) {
        int n = idx & 127;
        int k = idx >> 7;
        Wt[(size_t)n * K + k] = (_Float16)W[(size_t)k * 128 + n];
    }
}

// ---------------- WMMA GEMM: H[N x 128] = A[N x KD] * B[KD x 128] (f16 in, f32 out) ----------
// Block: 256 threads (8 waves). Block computes 128 rows x 128 cols.
// Full B panel staged once per block into LDS; A tile staged per k-step into a single LDS buffer.
// All global->LDS movement uses GLOBAL_LOAD_ASYNC_TO_LDS_B128 (ASYNCcnt).
template <int KD>
__global__ __launch_bounds__(256)
void k_gemm_wmma(const _Float16* __restrict__ A, const _Float16* __restrict__ Bt,
                 float* __restrict__ H, int Nrows) {
    constexpr int BSTRIDE = KD + 8;                 // 16B row pad -> rotate LDS banks
    __shared__ __align__(16) _Float16 As[128 * 32];
    __shared__ __align__(16) _Float16 Bs[128 * BSTRIDE];

    const int wave = threadIdx.x >> 5;     // 0..7
    const int lane = threadIdx.x & 31;
    const int m    = lane & 15;
    const int half = lane >> 4;            // 0 or 1
    const int blockRow = blockIdx.x * 128;

    // ---- stage entire B panel (128 cols x KD) into LDS, async ----
    {
        constexpr int NCH = 128 * KD / 8;          // 16-byte chunks
#pragma unroll 4
        for (int c = threadIdx.x; c < NCH; c += 256) {
            const int n  = c / (KD / 8);
            const int kc = (c % (KD / 8)) * 8;
            async_copy_b128((unsigned)(uintptr_t)&Bs[n * BSTRIDE + kc],
                            Bt + (size_t)n * KD + kc);
        }
    }

    v8f acc[8] = {};

    for (int k0 = 0; k0 < KD; k0 += 32) {
        // ---- stage A tile (128 rows x 32 k) into LDS, async ----
        {
            const int row = threadIdx.x >> 1;
            const int kk  = (threadIdx.x & 1) * 16;
            int grow = blockRow + row;
            if (grow >= Nrows) grow = Nrows - 1;   // clamp: garbage rows are never stored
            const unsigned loff = (unsigned)(uintptr_t)&As[row * 32 + kk];
            const _Float16* gp  = A + (size_t)grow * KD + k0 + kk;
            async_copy_b128(loff, gp);
            async_copy_b128(loff + 16u, gp + 8);
        }
        wait_async0();
        __syncthreads();

        // ---- A fragment per ISA 16-bit 16x32 layout ----
        // lane<16 : row m, K = {0..7, 16..23}; lane>=16: row m, K = {8..15, 24..31}
        FragH a;
        {
            const _Float16* abase = &As[(wave * 16 + m) * 32 + half * 8];
            a.q[0] = *reinterpret_cast<const uint4*>(abase);
            a.q[1] = *reinterpret_cast<const uint4*>(abase + 16);
        }

        // ---- preload all 8 B fragments, then run the WMMA chain back-to-back ----
        FragH b[8];
#pragma unroll
        for (int nt = 0; nt < 8; ++nt) {
            const _Float16* bp = &Bs[(nt * 16 + m) * BSTRIDE + k0 + half * 8];
            b[nt].q[0] = *reinterpret_cast<const uint4*>(bp);
            b[nt].q[1] = *reinterpret_cast<const uint4*>(bp + 16);
        }
#pragma unroll
        for (int nt = 0; nt < 8; ++nt) {
            acc[nt] = __builtin_amdgcn_wmma_f32_16x16x32_f16(
                /*neg_a=*/false, a.v, /*neg_b=*/false, b[nt].v,
                /*c_mod=*/(short)0, acc[nt], /*reuse_a=*/false, /*reuse_b=*/false);
        }
        __syncthreads();   // protect As (WAR) before next async stage
    }

    // ---- store D: element v -> (M = v + 8*half, N = nt*16 + m) ----
    const int rbase = blockRow + wave * 16;
#pragma unroll
    for (int nt = 0; nt < 8; ++nt) {
        const int col = nt * 16 + m;
#pragma unroll
        for (int v = 0; v < 8; ++v) {
            const int r = rbase + v + 8 * half;
            if (r < Nrows) H[(size_t)r * 128 + col] = acc[nt][v];
        }
    }
}

// ---------------- aggregation kernels ----------------
// O[i][c] = bias[c] + H[i][c] * dinv[i]^2   (self-loop, fully overwrites O)
__global__ void k_self_bias(const float* __restrict__ H, const float* __restrict__ bias,
                            const float* __restrict__ dinv, float* __restrict__ O, int N) {
    size_t i4 = (size_t)blockIdx.x * blockDim.x + threadIdx.x;   // over N*32 float4s
    if (i4 < (size_t)N * 32) {
        int node = (int)(i4 >> 5);
        int c0   = ((int)i4 & 31) * 4;
        float s  = dinv[node];
        s = s * s;
        float4 h = reinterpret_cast<const float4*>(H)[i4];
        float4 b = *reinterpret_cast<const float4*>(bias + c0);
        float4 o;
        o.x = b.x + h.x * s; o.y = b.y + h.y * s;
        o.z = b.z + h.z * s; o.w = b.w + h.w * s;
        reinterpret_cast<float4*>(O)[i4] = o;
    }
}

// one wave (32 lanes) per edge; lane owns 4 consecutive features (float4)
__global__ __launch_bounds__(256)
void k_scatter(const int* __restrict__ src, const int* __restrict__ dst,
               const float* __restrict__ norm, const float* __restrict__ H,
               float* __restrict__ O, int E) {
    int e = blockIdx.x * 8 + (threadIdx.x >> 5);
    if (e >= E) return;
    const int lane = threadIdx.x & 31;
    const int s = src[e];
    const int d = dst[e];
    const float nm = norm[e];
    float4 v = reinterpret_cast<const float4*>(H + (size_t)s * 128)[lane];
    float* op = O + (size_t)d * 128 + lane * 4;
    atomicAdd(op + 0, v.x * nm);
    atomicAdd(op + 1, v.y * nm);
    atomicAdd(op + 2, v.z * nm);
    atomicAdd(op + 3, v.w * nm);
}

__global__ void k_relu_to_f16(const float* __restrict__ O, _Float16* __restrict__ A16,
                              size_t n4) {
    size_t i = (size_t)blockIdx.x * blockDim.x + threadIdx.x;
    if (i < n4) {
        float4 v = reinterpret_cast<const float4*>(O)[i];
        _Float16* o = A16 + i * 4;
        o[0] = (_Float16)fmaxf(v.x, 0.0f);
        o[1] = (_Float16)fmaxf(v.y, 0.0f);
        o[2] = (_Float16)fmaxf(v.z, 0.0f);
        o[3] = (_Float16)fmaxf(v.w, 0.0f);
    }
}

__global__ void k_relu_f32(const float* __restrict__ O, float* __restrict__ out, size_t n4) {
    size_t i = (size_t)blockIdx.x * blockDim.x + threadIdx.x;
    if (i < n4) {
        float4 v = reinterpret_cast<const float4*>(O)[i];
        float4 o;
        o.x = fmaxf(v.x, 0.0f); o.y = fmaxf(v.y, 0.0f);
        o.z = fmaxf(v.z, 0.0f); o.w = fmaxf(v.w, 0.0f);
        reinterpret_cast<float4*>(out)[i] = o;
    }
}

// ---------------- launch ----------------
extern "C" void kernel_launch(void* const* d_in, const int* in_sizes, int n_in,
                              void* d_out, int out_size, void* d_ws, size_t ws_size,
                              hipStream_t stream) {
    const float* x    = (const float*)d_in[0];
    const int*   eidx = (const int*)d_in[1];
    const float* ew   = (const float*)d_in[2];
    const float* W1   = (const float*)d_in[3];
    const float* b1   = (const float*)d_in[4];
    const float* W2   = (const float*)d_in[5];
    const float* b2   = (const float*)d_in[6];
    const float* W3   = (const float*)d_in[7];
    const float* b3   = (const float*)d_in[8];

    const int N = in_sizes[0] / 256;   // 50000
    const int E = in_sizes[2];         // 800000
    const int* srcI = eidx;
    const int* dstI = eidx + E;

    // workspace carve-up (256B aligned)
    char* ws = (char*)d_ws;
    size_t off = 0;
    auto carve = [&](size_t bytes) -> char* {
        char* p = ws + off;
        off = (off + bytes + 255) & ~(size_t)255;
        return p;
    };
    float*    dinv = (float*)   carve((size_t)N * 4);
    float*    nrm  = (float*)   carve((size_t)E * 4);
    _Float16* X16  = (_Float16*)carve((size_t)N * 256 * 2);
    _Float16* A16  = (_Float16*)carve((size_t)N * 128 * 2);
    float*    H    = (float*)   carve((size_t)N * 128 * 4);
    float*    O    = (float*)   carve((size_t)N * 128 * 4);
    _Float16* Wt   = (_Float16*)carve((size_t)256 * 128 * 2);
    (void)ws_size;

    const int T = 256;
    const int gN    = (N + T - 1) / T;
    const int gE    = (E + T - 1) / T;
    const int gEw   = (E + 7) / 8;                 // wave-per-edge
    const int gGemm = (N + 127) / 128;
    const size_t nX4 = (size_t)N * 256 / 4;
    const size_t nF4 = (size_t)N * 128 / 4;
    const int gX4 = (int)((nX4 + T - 1) / T);
    const int gF4 = (int)((nF4 + T - 1) / T);

    // ---- normalization ----
    k_deg_init<<<gN, T, 0, stream>>>(dinv, N);
    k_deg_acc<<<gE, T, 0, stream>>>(dstI, ew, dinv, E);
    k_dinv<<<gN, T, 0, stream>>>(dinv, N);
    k_norm<<<gE, T, 0, stream>>>(srcI, dstI, ew, dinv, nrm, E);

    // ---- input to f16 ----
    k_f32_to_f16_v4<<<gX4, T, 0, stream>>>(x, X16, nX4);

    // ---- layer 1 (K=256) ----
    k_w_prep<<<(256 * 128 + T - 1) / T, T, 0, stream>>>(W1, Wt, 256);
    k_gemm_wmma<256><<<gGemm, T, 0, stream>>>(X16, Wt, H, N);
    k_self_bias<<<gF4, T, 0, stream>>>(H, b1, dinv, O, N);
    k_scatter<<<gEw, T, 0, stream>>>(srcI, dstI, nrm, H, O, E);
    k_relu_to_f16<<<gF4, T, 0, stream>>>(O, A16, nF4);

    // ---- layer 2 (K=128) ----
    k_w_prep<<<(128 * 128 + T - 1) / T, T, 0, stream>>>(W2, Wt, 128);
    k_gemm_wmma<128><<<gGemm, T, 0, stream>>>(A16, Wt, H, N);
    k_self_bias<<<gF4, T, 0, stream>>>(H, b2, dinv, O, N);
    k_scatter<<<gEw, T, 0, stream>>>(srcI, dstI, nrm, H, O, E);
    k_relu_to_f16<<<gF4, T, 0, stream>>>(O, A16, nF4);

    // ---- layer 3 (K=128) -> d_out ----
    k_w_prep<<<(128 * 128 + T - 1) / T, T, 0, stream>>>(W3, Wt, 128);
    k_gemm_wmma<128><<<gGemm, T, 0, stream>>>(A16, Wt, H, N);
    k_self_bias<<<gF4, T, 0, stream>>>(H, b3, dinv, O, N);
    k_scatter<<<gEw, T, 0, stream>>>(srcI, dstI, nrm, H, O, E);
    k_relu_f32<<<gF4, T, 0, stream>>>(O, (float*)d_out, nF4);
}
